// NeuralNetwork_7017976561936
// MI455X (gfx1250) — matrix operand order repaired and checked by reading, not compiled, not run on hardware
//
#include <hip/hip_runtime.h>

typedef __attribute__((ext_vector_type(16))) __bf16 v16bf;
typedef __attribute__((ext_vector_type(8)))  float  v8f;

union FragAB {
  v16bf v;
  struct { uint4 lo, hi; } q;
};

#define IMGD 144
#define CHD 16
#define MROWS 1296            // 36*36 conv1 output pixels
#define KD 48                 // 3ch * 4 * 4
#define FRAME_ELEMS (3*IMGD*IMGD)
#define FRAME_VEC4  (FRAME_ELEMS/4)      // 15552 float4 per frame

// K1 dynamic-LDS layout (bytes)
#define OFF_A    0
#define SZ_A     (MROWS*KD*2)            // 124416: bf16 im2col [m][k]
#define OFF_POOL (OFF_A + SZ_A)
#define SZ_POOL  (CHD*12*12*4)           // 9216: pooled1 as u32 bits
#define OFF_W2   (OFF_POOL + SZ_POOL)
#define SZ_W2    (256*4)                 // conv2 weights
#define OFF_Y2   (OFF_W2 + SZ_W2)
#define SZ_Y2    (16*4)
#define K1_LDS   (OFF_Y2 + SZ_Y2)        // ~134.7 KB (< 320 KB/WGP)

// ---------------------------------------------------------------------------
// Kernel 1: per-frame conv1(WMMA) + bn1 + relu + pool3 + conv2 + bn2 + relu +
// pool3  ->  c[frame] scalar.  One workgroup per frame (B*N = 2048).
// Roofline: x is 509 MB -> ~22us HBM floor; stream it exactly once as b128
// loads, keep every intermediate in LDS, conv1 via bf16 WMMA (f32 accum).
// ---------------------------------------------------------------------------
__global__ void __launch_bounds__(256)
frame_kernel(const float* __restrict__ X,
             const float* __restrict__ w1,
             const float* __restrict__ b1c,
             const float* __restrict__ g1, const float* __restrict__ bb1,
             const float* __restrict__ m1, const float* __restrict__ v1,
             const float* __restrict__ w2g,
             const float* __restrict__ b2c,
             const float* __restrict__ g2, const float* __restrict__ bb2,
             const float* __restrict__ m2, const float* __restrict__ v2,
             float* __restrict__ c_out)
{
  extern __shared__ char smem[];
  __bf16*   Alds = (__bf16*)(smem + OFF_A);
  unsigned* pool = (unsigned*)(smem + OFF_POOL);
  float*    w2l  = (float*)(smem + OFF_W2);
  float*    y2l  = (float*)(smem + OFF_Y2);

  const int tid   = threadIdx.x;
  const int lane  = tid & 31;
  const int wv    = tid >> 5;       // 0..7 (8 wave32 per WG)
  const int frame = blockIdx.x;

  // pooled init: relu output >= 0, so u32 bit-max with init 0 == float max-pool
  for (int i = tid; i < CHD*12*12; i += 256) pool[i] = 0u;
  w2l[tid] = w2g[tid];              // 256 threads, 256 weights

  // im2col stream: stride==kernel==4 so it is a pure permutation, and 4
  // consecutive x-pixels map to 4 consecutive k in the same m-row:
  //   global_load_b128 (f32x4) -> 4x bf16 -> ds_store_b64
  const float4* xf4 = (const float4*)(X + (size_t)frame * FRAME_ELEMS);
  for (int i4 = tid; i4 < FRAME_VEC4; i4 += 256) {
    __builtin_prefetch(xf4 + i4 + 8*256, 0, 1);   // global_prefetch_b8, ~32KB ahead
    float4 val = xf4[i4];
    int i   = i4 * 4;
    int ch  = i / (IMGD*IMGD);
    int rem = i - ch*(IMGD*IMGD);
    int yy  = rem / IMGD;
    int xx  = rem - yy*IMGD;              // multiple of 4
    int m   = (yy >> 2)*36 + (xx >> 2);
    int k   = ch*16 + (yy & 3)*4;         // xx&3 == 0
    union { __bf16 h[4]; uint2 u; } t;
    t.h[0] = (__bf16)val.x; t.h[1] = (__bf16)val.y;
    t.h[2] = (__bf16)val.z; t.h[3] = (__bf16)val.w;
    *(uint2*)(Alds + m*KD + k) = t.u;     // 8B aligned (k % 4 == 0)
  }

  // B fragments (conv1 weights as K x N), ISA dense-B striping:
// lanes 0-15: element e <-> K=e (e<8), K=8+e (e>=8) ; lanes 16-31: element e <-> K=8+e (e<8), K=16+e (e>=8) ; N = lane&15
  const int n  = lane & 15;
  const int kb = (lane < 16) ? 0 : 8;
  FragAB b0, b1;
  #pragma unroll
  for (int e = 0; e < 16; ++e) {
    int k0 = kb + e + ((e >= 8) ? 8 : 0);
    int k1 = 32 + k0;
    b0.v[e] = (__bf16)w1[n*KD + k0];
    b1.v[e] = (k1 < KD) ? (__bf16)w1[n*KD + k1] : (__bf16)0.0f;   // K 48..63 pad
  }

  // fused bias + batchnorm per output channel: relu(acc*alpha + beta)
  float alpha = g1[n] * rsqrtf(v1[n] + 1e-5f);
  float beta  = (b1c[n] - m1[n]) * alpha + bb1[n];

  __syncthreads();

  // conv1 GEMM: M=1296, N=16, K=48(pad 64) via v_wmma_f32_16x16x32_bf16
  for (int tile = wv; tile < MROWS/16; tile += 8) {
    const int  mrow   = tile*16 + (lane & 15);
    const char* abase = (const char*)(Alds + mrow*KD);   // 96 B row, 16B-aligned chunks
    FragAB a0, a1;
    if (lane < 16) {
      a0.q.lo = *(const uint4*)(abase + 0);    // K  0..7
      a0.q.hi = *(const uint4*)(abase + 32);   // K 16..23
      a1.q.lo = *(const uint4*)(abase + 64);   // K 32..39
    } else {
      a0.q.lo = *(const uint4*)(abase + 16);   // K  8..15
      a0.q.hi = *(const uint4*)(abase + 48);   // K 24..31
      a1.q.lo = *(const uint4*)(abase + 80);   // K 40..47
    }
    a1.q.hi = make_uint4(0u, 0u, 0u, 0u);      // K 48..63 zero pad

    v8f acc = {};
    acc = __builtin_amdgcn_wmma_f32_16x16x32_bf16(false, a0.v, false, b0.v,
                                                  (short)0, acc, false, false);
    acc = __builtin_amdgcn_wmma_f32_16x16x32_bf16(false, a1.v, false, b1.v,
                                                  (short)0, acc, false, false);

    // epilogue: bias+bn+relu, then 3x3/3 max-pool via ds_max_u32
    #pragma unroll
    for (int j = 0; j < 8; ++j) {
      int   m   = tile*16 + j + ((lane < 16) ? 0 : 8);   // C/D tile M mapping
      float val = fmaxf(0.0f, acc[j]*alpha + beta);
      int oy = m / 36, ox = m - oy*36;
      atomicMax(&pool[n*144 + (oy/3)*12 + (ox/3)], __float_as_uint(val));
    }
  }

  __syncthreads();

  // conv2: 16ch 4x4 s4 over 12x12 -> 3x3, + bias + bn2 + relu  (tiny)
  if (tid < 9) {
    int oy = tid / 3, ox = tid - oy*3;
    float s = 0.0f;
    for (int ch = 0; ch < 16; ++ch)
      for (int ky = 0; ky < 4; ++ky)
        for (int kx = 0; kx < 4; ++kx)
          s += __uint_as_float(pool[ch*144 + (oy*4+ky)*12 + (ox*4+kx)])
             * w2l[ch*16 + ky*4 + kx];
    float s2 = g2[0] * rsqrtf(v2[0] + 1e-5f);
    float yv = (s + b2c[0] - m2[0]) * s2 + bb2[0];
    y2l[tid] = fmaxf(0.0f, yv);
  }
  __syncthreads();
  if (tid == 0) {                      // 3x3/3 pool of a 3x3 map == max of 9
    float mx = 0.0f;
    for (int i = 0; i < 9; ++i) mx = fmaxf(mx, y2l[i]);
    c_out[frame] = mx;
  }
}

// ---------------------------------------------------------------------------
// Kernel 2: gating MLP per sample; argmax(softmax(z)) == argmax(z)
// ---------------------------------------------------------------------------
__global__ void gate_kernel(const float* __restrict__ c,
                            const float* __restrict__ pw1, const float* __restrict__ pb1,
                            const float* __restrict__ pw2, const float* __restrict__ pb2,
                            const float* __restrict__ pw3, const float* __restrict__ pb3,
                            int* __restrict__ p)
{
  __shared__ float h1[32], h2[32], lg[6];
  const int b = blockIdx.x, j = threadIdx.x;
  const float* cr = c + b*64;
  float s = pb1[j];
  for (int k = 0; k < 64; ++k) s += cr[k] * pw1[j*64 + k];
  h1[j] = tanhf(s);
  __syncthreads();
  s = pb2[j];
  for (int k = 0; k < 32; ++k) s += h1[k] * pw2[j*32 + k];
  h2[j] = tanhf(s);
  __syncthreads();
  if (j < 6) {
    s = pb3[j];
    for (int k = 0; k < 32; ++k) s += h2[k] * pw3[j*32 + k];
    lg[j] = s;
  }
  __syncthreads();
  if (j == 0) {
    int best = 0; float bv = lg[0];
    for (int a = 1; a < 6; ++a) if (lg[a] > bv) { bv = lg[a]; best = a; }
    p[b] = best;
  }
}

// ---------------------------------------------------------------------------
// Kernel 3: per-expert LSTM chain.  hn_tab chaining across the batch is only
// sequential *within* an expert -> 6 independent chains, one WG each.
// 128 lanes = one gate row each (whh row resident in VGPRs); h,c in LDS.
// ---------------------------------------------------------------------------
__global__ void lstm_kernel(const float* __restrict__ c, const int* __restrict__ p,
                            const float* __restrict__ wih, const float* __restrict__ whh,
                            const float* __restrict__ bih, const float* __restrict__ bhh,
                            const float* __restrict__ hn0, float* __restrict__ r)
{
  __shared__ float hh[32], cc[32], gg[128];
  const int e = blockIdx.x;      // expert 0..5
  const int l = threadIdx.x;     // gate row 0..127
  float wrow[32];
  #pragma unroll
  for (int k = 0; k < 32; ++k) wrow[k] = whh[(e*128 + l)*32 + k];
  const float wi = wih[e*128 + l];
  const float bs = bih[e*128 + l] + bhh[e*128 + l];
  if (l < 32) hh[l] = hn0[e*32 + l];
  __syncthreads();

  for (int b = 0; b < 32; ++b) {
    if (p[b] != e) continue;               // uniform across the WG
    if (l < 32) cc[l] = 0.0f;
    __syncthreads();
    for (int t = 0; t < 64; ++t) {
      float xt = c[b*64 + t];
      float g = wi*xt + bs;
      for (int k = 0; k < 32; ++k) g += wrow[k] * hh[k];
      gg[l] = g;
      __syncthreads();
      if (l < 32) {
        float ig = gg[l], fg = gg[32+l], gv = gg[64+l], og = gg[96+l];
        float c2 = (1.f/(1.f+expf(-fg)))*cc[l] + (1.f/(1.f+expf(-ig)))*tanhf(gv);
        float h2 = (1.f/(1.f+expf(-og)))*tanhf(c2);
        cc[l] = c2; hh[l] = h2;
        if (l == 31) r[b*64 + t] = h2;     // outs = hh2[-1]
      }
      __syncthreads();
    }
  }
}

// ---------------------------------------------------------------------------
// Kernel 4: out = r @ out_w.T + out_b   -> [32, 6]
// ---------------------------------------------------------------------------
__global__ void out_kernel(const float* __restrict__ r, const float* __restrict__ ow,
                           const float* __restrict__ ob, float* __restrict__ out)
{
  int i = blockIdx.x*blockDim.x + threadIdx.x;
  if (i >= 32*6) return;
  int b = i / 6, a = i - b*6;
  float s = ob[a];
  for (int t = 0; t < 64; ++t) s += r[b*64 + t] * ow[a*64 + t];
  out[i] = s;
}

extern "C" void kernel_launch(void* const* d_in, const int* in_sizes, int n_in,
                              void* d_out, int out_size, void* d_ws, size_t ws_size,
                              hipStream_t stream)
{
  const float* X       = (const float*)d_in[0];
  const float* conv1_w = (const float*)d_in[1];
  const float* conv1_b = (const float*)d_in[2];
  const float* bn1_g   = (const float*)d_in[3];
  const float* bn1_b   = (const float*)d_in[4];
  const float* bn1_m   = (const float*)d_in[5];
  const float* bn1_v   = (const float*)d_in[6];
  const float* conv2_w = (const float*)d_in[7];
  const float* conv2_b = (const float*)d_in[8];
  const float* bn2_g   = (const float*)d_in[9];
  const float* bn2_b   = (const float*)d_in[10];
  const float* bn2_m   = (const float*)d_in[11];
  const float* bn2_v   = (const float*)d_in[12];
  const float* pre_w1  = (const float*)d_in[13];
  const float* pre_b1  = (const float*)d_in[14];
  const float* pre_w2  = (const float*)d_in[15];
  const float* pre_b2  = (const float*)d_in[16];
  const float* pre_w3  = (const float*)d_in[17];
  const float* pre_b3  = (const float*)d_in[18];
  const float* lstm_wih = (const float*)d_in[19];
  const float* lstm_whh = (const float*)d_in[20];
  const float* lstm_bih = (const float*)d_in[21];
  const float* lstm_bhh = (const float*)d_in[22];
  const float* hn0     = (const float*)d_in[23];
  const float* out_w   = (const float*)d_in[24];
  const float* out_b   = (const float*)d_in[25];

  float* c_ws = (float*)d_ws;                                          // [2048]
  int*   p_ws = (int*)((char*)d_ws + 2048*sizeof(float));              // [32]
  float* r_ws = (float*)((char*)d_ws + 2048*sizeof(float) + 32*sizeof(int)); // [2048]

  frame_kernel<<<32*64, 256, K1_LDS, stream>>>(X, conv1_w, conv1_b,
      bn1_g, bn1_b, bn1_m, bn1_v, conv2_w, conv2_b, bn2_g, bn2_b, bn2_m, bn2_v,
      c_ws);
  gate_kernel<<<32, 32, 0, stream>>>(c_ws, pre_w1, pre_b1, pre_w2, pre_b2,
                                     pre_w3, pre_b3, p_ws);
  lstm_kernel<<<6, 128, 0, stream>>>(c_ws, p_ws, lstm_wih, lstm_whh,
                                     lstm_bih, lstm_bhh, hn0, r_ws);
  out_kernel<<<1, 192, 0, stream>>>(r_ws, out_w, out_b, (float*)d_out);
}
